// SpaCross_model_352187318913
// MI455X (gfx1250) — matrix operand order repaired and checked by reading, not compiled, not run on hardware
//
#include <hip/hip_runtime.h>
#include <hip/hip_bf16.h>

typedef __attribute__((ext_vector_type(16))) __bf16 v16bf;
typedef __attribute__((ext_vector_type(8)))  float  v8f;

union FragU {
    uint4 q[2];
    v16bf v;
};

__device__ __forceinline__ unsigned short f2bf(float f) {
    unsigned int x = __float_as_uint(f);
    unsigned int r = x + 0x7FFFu + ((x >> 16) & 1u);
    return (unsigned short)(r >> 16);
}

// ---------------------------------------------------------------- utilities
__global__ void k_fill_f32(float* p, float v, size_t n) {
    size_t i = (size_t)blockIdx.x * blockDim.x + threadIdx.x;
    if (i < n) p[i] = v;
}
__global__ void k_f32_to_bf16(const float* __restrict__ in, unsigned short* __restrict__ out, size_t n) {
    size_t i = (size_t)blockIdx.x * blockDim.x + threadIdx.x;
    if (i < n) out[i] = f2bf(in[i]);
}
// weights: f32 [K x Nc] -> bf16 transposed [Nc x K] so B fragments are K-contiguous
__global__ void k_f32_to_bf16_T(const float* __restrict__ in, unsigned short* __restrict__ out,
                                int K, int Nc) {
    size_t i = (size_t)blockIdx.x * blockDim.x + threadIdx.x;
    if (i >= (size_t)K * Nc) return;
    int k = (int)(i / Nc), n = (int)(i % Nc);
    out[(size_t)n * K + k] = f2bf(in[i]);
}

// ---------------------------------------------------------------- GCN norm
__global__ void k_deg_accum(const int* __restrict__ dst, float* deg, int E) {
    int e = blockIdx.x * blockDim.x + threadIdx.x;
    if (e < E) atomicAdd(&deg[dst[e]], 1.0f);
}
__global__ void k_dinv(const float* __restrict__ deg, float* __restrict__ dinv, int N) {
    int i = blockIdx.x * blockDim.x + threadIdx.x;
    if (i < N) { float d = deg[i]; dinv[i] = d > 0.0f ? rsqrtf(d) : 0.0f; }
}
__global__ void k_edge_norm(const int* __restrict__ src, const int* __restrict__ dst,
                            const float* __restrict__ dinv, float* __restrict__ en, int E) {
    int e = blockIdx.x * blockDim.x + threadIdx.x;
    if (e < E) en[e] = dinv[src[e]] * dinv[dst[e]];
}

// ---------------------------------------------------------------- masking
__global__ void k_set_rows_token(unsigned short* __restrict__ mat, const int* __restrict__ rows,
                                 int nrows, const float* __restrict__ tok, int C) {
    size_t i = (size_t)blockIdx.x * blockDim.x + threadIdx.x;
    if (i >= (size_t)nrows * C) return;
    int j = (int)(i / C), f = (int)(i % C);
    mat[(size_t)rows[j] * C + f] = f2bf(tok[f]);
}
__global__ void k_set_rows_gather(unsigned short* __restrict__ mat, const int* __restrict__ rows,
                                  const int* __restrict__ srcs, int nrows,
                                  const float* __restrict__ x, int C) {
    size_t i = (size_t)blockIdx.x * blockDim.x + threadIdx.x;
    if (i >= (size_t)nrows * C) return;
    int j = (int)(i / C), f = (int)(i % C);
    mat[(size_t)rows[j] * C + f] = f2bf(x[(size_t)srcs[j] * C + f]);
}
__global__ void k_zero_rows(float* __restrict__ mat, const int* __restrict__ rows, int nrows, int C) {
    size_t i = (size_t)blockIdx.x * blockDim.x + threadIdx.x;
    if (i >= (size_t)nrows * C) return;
    int j = (int)(i / C), f = (int)(i % C);
    mat[(size_t)rows[j] * C + f] = 0.0f;
}

// ---------------------------------------------------------------- WMMA GEMM
// C[M,Nc] = A[M,K](bf16 row-major) * Bt[Nc,K](bf16, pre-transposed) (+ bias).
// 8 waves/block; each wave computes a 16x32 output strip: one A fragment feeds
// two v_wmma_f32_16x16x32_bf16 per K-step. All operands are 16B global loads
// that land directly in a contiguous k = 16*half + s fragment order (no LDS, no shuffles).
__global__ void k_gemm_wmma(const unsigned short* __restrict__ A,
                            const unsigned short* __restrict__ Bt,
                            const float* __restrict__ bias,
                            float* __restrict__ C,
                            int M, int Nc, int K, int tilesN2, int tilesTotal) {
    int wave = threadIdx.x >> 5;
    int lane = threadIdx.x & 31;
    int tileId = blockIdx.x * (blockDim.x >> 5) + wave;
    if (tileId >= tilesTotal) return;          // wave-uniform: EXEC stays all-ones
    int tm = tileId / tilesN2;
    int tn = (tileId % tilesN2) * 2;

    int half = lane >> 4;                       // 0: lanes 0-15, 1: lanes 16-31
    int col  = lane & 15;
    int khiA = half * 16;                       // A: per-half-wave K base
    int khiB = half * 16;                       // B: per-half-wave K base

    const unsigned short* Arow  = A  + (size_t)(tm * 16 + col) * K;
    const unsigned short* Brow0 = Bt + (size_t)(tn * 16 + col) * K + khiB;
    const unsigned short* Brow1 = Bt + (size_t)((tn + 1) * 16 + col) * K + khiB;

    v8f acc0 = {}, acc1 = {};
#pragma unroll 2
    for (int k0 = 0; k0 < K; k0 += 32) {
        FragU a, b0, b1;
        a.q[0]  = *(const uint4*)(Arow + k0 + khiA);        // K = khiA..khiA+7
        a.q[1]  = *(const uint4*)(Arow + k0 + khiA + 8);    // K = khiA+8..+15
        b0.q[0] = *(const uint4*)(Brow0 + k0);              // K = khiB..khiB+7
        b0.q[1] = *(const uint4*)(Brow0 + k0 + 8);          // K = khiB+8..+15
        b1.q[0] = *(const uint4*)(Brow1 + k0);
        b1.q[1] = *(const uint4*)(Brow1 + k0 + 8);
        acc0 = __builtin_amdgcn_wmma_f32_16x16x32_bf16(false, a.v, false, b0.v, (short)0, acc0, false, false);
        acc1 = __builtin_amdgcn_wmma_f32_16x16x32_bf16(false, a.v, false, b1.v, (short)0, acc1, false, false);
    }

    int cm  = tm * 16 + half * 8;
    int cn0 = tn * 16 + col;
    int cn1 = cn0 + 16;
    float ba0 = bias ? bias[cn0] : 0.0f;
    float ba1 = bias ? bias[cn1] : 0.0f;
#pragma unroll
    for (int r = 0; r < 8; ++r) {
        C[(size_t)(cm + r) * Nc + cn0] = acc0[r] + ba0;
        C[(size_t)(cm + r) * Nc + cn1] = acc1[r] + ba1;
    }
}

// ---------------------------------------------------------------- BatchNorm
__global__ void k_bn_stats(const float* __restrict__ X, int M, int C,
                           float* __restrict__ mean, float* __restrict__ var) {
    __shared__ float s1[256], s2[256];
    int c = blockIdx.x, t = threadIdx.x;
    float s = 0.0f, q = 0.0f;
    for (int r = t; r < M; r += 256) {
        float v = X[(size_t)r * C + c];
        s += v; q += v * v;
    }
    s1[t] = s; s2[t] = q;
    __syncthreads();
    for (int o = 128; o > 0; o >>= 1) {
        if (t < o) { s1[t] += s1[t + o]; s2[t] += s2[t + o]; }
        __syncthreads();
    }
    if (t == 0) {
        float m = s1[0] / (float)M;
        mean[c] = m;
        var[c]  = s2[0] / (float)M - m * m;
    }
}
__global__ void k_bn_apply(const float* __restrict__ X, const float* __restrict__ mean,
                           const float* __restrict__ var, const float* __restrict__ g,
                           const float* __restrict__ beta, float eps, int act,
                           float* __restrict__ outF, unsigned short* __restrict__ outB,
                           size_t total, int cmask) {
    size_t i = (size_t)blockIdx.x * blockDim.x + threadIdx.x;
    if (i >= total) return;
    int c = (int)(i & (size_t)cmask);
    float h = g[c] * (X[i] - mean[c]) * rsqrtf(var[c] + eps) + beta[c];
    if (act == 1) h = h > 0.0f ? h : (expf(h) - 1.0f);   // ELU
    else if (act == 2) h = h > 0.0f ? h : 0.0f;          // ReLU
    if (outF) outF[i] = h;
    if (outB) outB[i] = f2bf(h);
}

// ---------------------------------------------------------------- GCN scatter
__global__ void k_rows_bias_init(float* __restrict__ agg, const float* __restrict__ b,
                                 size_t total, int cmask) {
    size_t i = (size_t)blockIdx.x * blockDim.x + threadIdx.x;
    if (i < total) agg[i] = b[(int)(i & (size_t)cmask)];
}
__global__ void k_scatter_edges(const float* __restrict__ P, const int* __restrict__ src,
                                const int* __restrict__ dst, const float* __restrict__ en,
                                float* __restrict__ agg, size_t total, int logC, int cmask) {
    size_t i = (size_t)blockIdx.x * blockDim.x + threadIdx.x;
    if (i >= total) return;
    int e = (int)(i >> logC), f = (int)(i & (size_t)cmask);
    int s = src[e], d = dst[e];
    atomicAdd(&agg[((size_t)d << logC) + f], P[((size_t)s << logC) + f] * en[e]);
}
__global__ void k_self_loops(const float* __restrict__ P, const float* __restrict__ dinv,
                             float* __restrict__ agg, size_t total, int logC, int cmask) {
    size_t i = (size_t)blockIdx.x * blockDim.x + threadIdx.x;
    if (i >= total) return;
    int nd = (int)(i >> logC);
    float dv = dinv[nd];
    agg[i] += P[i] * dv * dv;
}

// ---------------------------------------------------------------- losses
__global__ void k_loss_pair(const float* __restrict__ S, const float* __restrict__ T,
                            const int* __restrict__ ia, const int* __restrict__ ib,
                            int n, int mode, float* acc) {
    int w = (int)((blockIdx.x * blockDim.x + threadIdx.x) >> 5);
    int lane = threadIdx.x & 31;
    if (w >= n) return;
    const float* sp = S + (size_t)ia[w] * 64;
    const float* tp = T + (size_t)ib[w] * 64;
    float d = 0.0f, ss = 0.0f, tt = 0.0f;
    for (int f = lane; f < 64; f += 32) {
        float a = sp[f], b = tp[f];
        d += a * b; ss += a * a; tt += b * b;
    }
    for (int o = 16; o > 0; o >>= 1) {
        d += __shfl_xor(d, o, 32); ss += __shfl_xor(ss, o, 32); tt += __shfl_xor(tt, o, 32);
    }
    if (lane == 0) {
        float cs = d * rsqrtf(ss * tt + 1e-30f);
        float q  = 0.5f * (1.0f + cs);
        float val = (mode == 0) ? (-2.0f * logf(fmaxf(q, 1e-30f)))
                                : (-logf(fmaxf(1.0f - q * q, 1e-30f)));
        atomicAdd(acc + mode, val);
    }
}
__global__ void k_loss_rec(const float* __restrict__ R, const float* __restrict__ X,
                           const int* __restrict__ rows, int n, float* acc) {
    int w = (int)((blockIdx.x * blockDim.x + threadIdx.x) >> 5);
    int lane = threadIdx.x & 31;
    if (w >= n) return;
    int i = rows[w];
    const float* rp = R + (size_t)i * 512;
    const float* xp = X + (size_t)i * 512;
    float d = 0.0f, ss = 0.0f, tt = 0.0f;
    for (int f = lane; f < 512; f += 32) {
        float a = rp[f], b = xp[f];
        d += a * b; ss += a * a; tt += b * b;
    }
    for (int o = 16; o > 0; o >>= 1) {
        d += __shfl_xor(d, o, 32); ss += __shfl_xor(ss, o, 32); tt += __shfl_xor(tt, o, 32);
    }
    if (lane == 0) {
        float cs = d * rsqrtf(ss * tt + 1e-30f);
        float q  = 0.5f * (1.0f + cs);
        atomicAdd(acc + 2, -2.0f * logf(fmaxf(q, 1e-30f)));
    }
}
__global__ void k_finalize(const float* __restrict__ acc, float* __restrict__ out, float invNM) {
    out[0] = 0.5f * (acc[0] + acc[1]) * invNM;
    out[1] = acc[2] * invNM;
    out[2] = 0.0f;
}

// ================================================================ host
static inline unsigned int cdiv(size_t a, size_t b) { return (unsigned int)((a + b - 1) / b); }

extern "C" void kernel_launch(void* const* d_in, const int* in_sizes, int n_in,
                              void* d_out, int out_size, void* d_ws, size_t ws_size,
                              hipStream_t stream) {
    const int D_IN = 512, H1 = 256, H2 = 128, GH = 128, LAT = 64;
    const int N  = in_sizes[0] / D_IN;
    const int E  = in_sizes[1] / 2;
    const int NM = in_sizes[2];
    const int NTOK = in_sizes[4];
    const int NNOI = in_sizes[5];

    const float* x    = (const float*)d_in[0];
    const int*   src  = (const int*)d_in[1];
    const int*   dst  = src + E;
    const int* maskN  = (const int*)d_in[2];
    const int* keepN  = (const int*)d_in[3];
    const int* tokN   = (const int*)d_in[4];
    const int* noiN   = (const int*)d_in[5];
    const int* noiS   = (const int*)d_in[6];
    const int* negx   = (const int*)d_in[7];
    const int* negy   = (const int*)d_in[8];
    const int PO = 9, PT = 25;                 // 16 params each: {l1,l2,g1,g2} x {w,b,g,beta}
    const float* e2d_w   = (const float*)d_in[41];
    const float* proj_w  = (const float*)d_in[42];
    const float* proj_b  = (const float*)d_in[43];
    const float* proj_g  = (const float*)d_in[44];
    const float* proj_be = (const float*)d_in[45];
    const float* dec_w   = (const float*)d_in[46];
    const float* dec_b   = (const float*)d_in[47];
    const float* mtok    = (const float*)d_in[48];

    // ---- workspace bump allocator
    size_t off = 0;
    auto alloc = [&](size_t bytes) -> void* {
        void* p = (char*)d_ws + off;
        off += (bytes + 255) & ~(size_t)255;
        return p;
    };
    float* deg    = (float*)alloc((size_t)N * 4);
    float* dinvv  = (float*)alloc((size_t)N * 4);
    float* enorm  = (float*)alloc((size_t)E * 4);
    unsigned short* wb1 = (unsigned short*)alloc((size_t)D_IN * H1 * 2);
    unsigned short* wb2 = (unsigned short*)alloc((size_t)H1 * H2 * 2);
    unsigned short* wb3 = (unsigned short*)alloc((size_t)H2 * GH * 2);
    unsigned short* wb4 = (unsigned short*)alloc((size_t)GH * LAT * 2);
    unsigned short* whe = (unsigned short*)alloc((size_t)LAT * 64 * 2);
    unsigned short* whp = (unsigned short*)alloc((size_t)64 * LAT * 2);
    unsigned short* whd = (unsigned short*)alloc((size_t)LAT * D_IN * 2);
    unsigned short* inBf   = (unsigned short*)alloc((size_t)N * D_IN * 2);
    float*          h1f    = (float*)alloc((size_t)N * H1 * 4);
    unsigned short* h1bf   = (unsigned short*)alloc((size_t)N * H1 * 2);
    float*          h2f    = (float*)alloc((size_t)N * H2 * 4);
    unsigned short* h2bf   = (unsigned short*)alloc((size_t)N * H2 * 2);
    float*          Pbuf   = (float*)alloc((size_t)N * GH * 4);
    float*          agg    = (float*)alloc((size_t)N * GH * 4);
    float*          encrep = (float*)alloc((size_t)N * LAT * 4);
    unsigned short* encbf  = (unsigned short*)alloc((size_t)N * LAT * 2);
    float*          rept   = (float*)alloc((size_t)N * LAT * 4);
    float*          rep0   = (float*)alloc((size_t)N * LAT * 4);
    unsigned short* rep0bf = (unsigned short*)alloc((size_t)N * LAT * 2);
    float*          repf   = (float*)alloc((size_t)N * LAT * 4);
    unsigned short* repbf  = (unsigned short*)alloc((size_t)N * LAT * 2);
    float*          recon  = (float*)alloc((size_t)N * D_IN * 4);
    float*          meanb  = (float*)alloc(512 * 4);
    float*          varb   = (float*)alloc(512 * 4);
    float*          acc    = (float*)alloc(4 * 4);
    (void)ws_size; (void)n_in; (void)out_size;

    auto cvt = [&](const float* s, unsigned short* d, size_t n) {
        k_f32_to_bf16<<<cdiv(n, 256), 256, 0, stream>>>(s, d, n);
    };
    auto cvtT = [&](const float* s, unsigned short* d, int K, int Nc) {
        k_f32_to_bf16_T<<<cdiv((size_t)K * Nc, 256), 256, 0, stream>>>(s, d, K, Nc);
    };
    auto gemm = [&](const unsigned short* A, const unsigned short* Bt, const float* bias,
                    float* C, int M, int Nc, int K) {
        int tilesN2 = Nc / 32, tiles = (M / 16) * tilesN2;
        k_gemm_wmma<<<cdiv(tiles, 8), 256, 0, stream>>>(A, Bt, bias, C, M, Nc, K, tilesN2, tiles);
    };
    auto bn = [&](float* X, int C, const float* g, const float* beta, float eps, int act,
                  float* outF, unsigned short* outB) {
        k_bn_stats<<<C, 256, 0, stream>>>(X, N, C, meanb, varb);
        size_t tot = (size_t)N * C;
        k_bn_apply<<<cdiv(tot, 256), 256, 0, stream>>>(X, meanb, varb, g, beta, eps, act,
                                                       outF, outB, tot, C - 1);
    };
    auto gcn = [&](const unsigned short* hbf, const unsigned short* wbf, const float* b,
                   int Cin, int Cout) {
        gemm(hbf, wbf, nullptr, Pbuf, N, Cout, Cin);
        int logC = (Cout == 128) ? 7 : 6;
        size_t totN = (size_t)N * Cout, totE = (size_t)E * Cout;
        k_rows_bias_init<<<cdiv(totN, 256), 256, 0, stream>>>(agg, b, totN, Cout - 1);
        k_scatter_edges<<<cdiv(totE, 256), 256, 0, stream>>>(Pbuf, src, dst, enorm, agg,
                                                             totE, logC, Cout - 1);
        k_self_loops<<<cdiv(totN, 256), 256, 0, stream>>>(Pbuf, dinvv, agg, totN, logC, Cout - 1);
    };
    auto encoder = [&](int pb, float* outRep, unsigned short* outRepBf) {
        cvtT((const float*)d_in[pb + 0],  wb1, D_IN, H1);
        cvtT((const float*)d_in[pb + 4],  wb2, H1, H2);
        cvtT((const float*)d_in[pb + 8],  wb3, H2, GH);
        cvtT((const float*)d_in[pb + 12], wb4, GH, LAT);
        gemm(inBf, wb1, (const float*)d_in[pb + 1], h1f, N, H1, D_IN);
        bn(h1f, H1, (const float*)d_in[pb + 2], (const float*)d_in[pb + 3], 1e-3f, 1, nullptr, h1bf);
        gemm(h1bf, wb2, (const float*)d_in[pb + 5], h2f, N, H2, H1);
        bn(h2f, H2, (const float*)d_in[pb + 6], (const float*)d_in[pb + 7], 1e-3f, 1, nullptr, h2bf);
        gcn(h2bf, wb3, (const float*)d_in[pb + 9], H2, GH);
        bn(agg, GH, (const float*)d_in[pb + 10], (const float*)d_in[pb + 11], 1e-5f, 2, nullptr, h2bf);
        gcn(h2bf, wb4, (const float*)d_in[pb + 13], GH, LAT);
        bn(agg, LAT, (const float*)d_in[pb + 14], (const float*)d_in[pb + 15], 1e-5f, 0, outRep, outRepBf);
    };

    // ---- 1. GCN normalization (self-loops fold into deg init = 1)
    k_fill_f32<<<cdiv((size_t)N, 256), 256, 0, stream>>>(deg, 1.0f, (size_t)N);
    k_deg_accum<<<cdiv((size_t)E, 256), 256, 0, stream>>>(dst, deg, E);
    k_dinv<<<cdiv((size_t)N, 256), 256, 0, stream>>>(deg, dinvv, N);
    k_edge_norm<<<cdiv((size_t)E, 256), 256, 0, stream>>>(src, dst, dinvv, enorm, E);

    // ---- 2. masked online input (bf16), online encoder
    cvt(x, inBf, (size_t)N * D_IN);
    k_set_rows_token<<<cdiv((size_t)NTOK * D_IN, 256), 256, 0, stream>>>(inBf, tokN, NTOK, mtok, D_IN);
    k_set_rows_gather<<<cdiv((size_t)NNOI * D_IN, 256), 256, 0, stream>>>(inBf, noiN, noiS, NNOI, x, D_IN);
    encoder(PO, encrep, encbf);

    // ---- 3. target input (keep nodes -> mask token), target encoder
    cvt(x, inBf, (size_t)N * D_IN);
    k_set_rows_token<<<cdiv((size_t)(N - NM) * D_IN, 256), 256, 0, stream>>>(inBf, keepN, N - NM, mtok, D_IN);
    encoder(PT, rept, nullptr);

    // ---- 4. head: e2d -> zero masked rows -> GCN proj + BN -> decode
    cvtT(e2d_w, whe, LAT, 64);
    cvtT(proj_w, whp, 64, LAT);
    cvtT(dec_w, whd, LAT, D_IN);
    gemm(encbf, whe, nullptr, rep0, N, 64, LAT);
    k_zero_rows<<<cdiv((size_t)NM * 64, 256), 256, 0, stream>>>(rep0, maskN, NM, 64);
    cvt(rep0, rep0bf, (size_t)N * 64);
    gcn(rep0bf, whp, proj_b, 64, LAT);
    bn(agg, LAT, proj_g, proj_be, 1e-5f, 0, repf, repbf);
    gemm(repbf, whd, dec_b, recon, N, D_IN, LAT);

    // ---- 5. losses
    k_fill_f32<<<1, 4, 0, stream>>>(acc, 0.0f, 4);
    unsigned int lb = cdiv((size_t)NM * 32, 256);
    k_loss_pair<<<lb, 256, 0, stream>>>(repf, rept, maskN, maskN, NM, 0, acc);
    k_loss_pair<<<lb, 256, 0, stream>>>(repf, rept, negx, negy, NM, 1, acc);
    k_loss_rec<<<lb, 256, 0, stream>>>(recon, x, maskN, NM, acc);
    k_finalize<<<1, 1, 0, stream>>>(acc, (float*)d_out, 1.0f / (float)NM);
}
